// MultiHeadAttention_26414048870703
// MI455X (gfx1250) — matrix operand order repaired and checked by reading, not compiled, not run on hardware
//
#include <hip/hip_runtime.h>
#include <math.h>

// ---------------------------------------------------------------------------
// MI455X (gfx1250) multi-head attention forward.
// bf16 WMMA (v_wmma_f32_16x16x32_bf16) for QKV GEMM, attention, output GEMM.
// fp32 accumulation + fp32 softmax/RoPE. wave32, 256-thread blocks (8 waves).
// V staged transposed in LDS so PV B-fragments are contiguous ds_load_b128.
// global_prefetch_b8 issued for next tiles to overlap HBM latency with WMMA.
// ---------------------------------------------------------------------------

typedef __attribute__((ext_vector_type(16))) __bf16 v16bf;
typedef __attribute__((ext_vector_type(8)))  float  v8f;

__device__ __forceinline__ __bf16 to_bf16(float f) {
  unsigned u = __builtin_bit_cast(unsigned, f);
  unsigned r = u + 0x7FFFu + ((u >> 16) & 1u);     // round-to-nearest-even
  unsigned short h = (unsigned short)(r >> 16);
  return __builtin_bit_cast(__bf16, h);
}
__device__ __forceinline__ float bf16_to_f32(__bf16 b) {
  unsigned short h = __builtin_bit_cast(unsigned short, b);
  unsigned u = ((unsigned)h) << 16;
  return __builtin_bit_cast(float, u);
}

__device__ __forceinline__ v8f wmma_bf16(v16bf a, v16bf b, v8f c) {
  return __builtin_amdgcn_wmma_f32_16x16x32_bf16(
      /*neg_a=*/false, a, /*neg_b=*/false, b,
      /*c_mod=*/(short)0, c, /*reuse_a=*/false, /*reuse_b=*/false);
}

// ---------------------------------------------------------------------------
// fp32 -> bf16 elementwise convert (grid-stride)
// ---------------------------------------------------------------------------
__global__ __launch_bounds__(256) void cvt_f32_bf16(const float* __restrict__ in,
                                                    __bf16* __restrict__ out,
                                                    size_t n) {
  size_t i = (size_t)blockIdx.x * blockDim.x + threadIdx.x;
  size_t stride = (size_t)gridDim.x * blockDim.x;
  for (; i < n; i += stride) out[i] = to_bf16(in[i]);
}

// ---------------------------------------------------------------------------
// C[m][n] = sum_k A[m][k] * W[n][k]   (X @ W^T), bf16 inputs, f32 accum.
// Tile 128x128, K-step 32. 8 waves: each wave owns 32(M) x 64(N) = 2x4 frags.
// M, N, K must be multiples of 128/128/32 (true for this problem).
// ---------------------------------------------------------------------------
template <bool OUT_BF16>
__global__ __launch_bounds__(256)
void gemm_xwt(const __bf16* __restrict__ A, const __bf16* __restrict__ W,
              void* __restrict__ Cout, int M, int N, int K) {
  __shared__ __align__(16) __bf16 As[128][40];  // 32 cols + 8 pad
  __shared__ __align__(16) __bf16 Bs[128][40];

  const int tid    = threadIdx.x;
  const int lane   = tid & 31;
  const int wave   = tid >> 5;
  const int laneLo = lane & 15;
  const int hi     = lane >> 4;
  const int wm     = (wave & 3) * 32;
  const int wn     = (wave >> 2) * 64;
  const int m0     = blockIdx.y * 128;
  const int n0     = blockIdx.x * 128;

  v8f acc[2][4];
#pragma unroll
  for (int mi = 0; mi < 2; ++mi)
#pragma unroll
    for (int ni = 0; ni < 4; ++ni)
      acc[mi][ni] = (v8f){0.f, 0.f, 0.f, 0.f, 0.f, 0.f, 0.f, 0.f};

  for (int k0 = 0; k0 < K; k0 += 32) {
#pragma unroll
    for (int i = 0; i < 2; ++i) {
      int l    = tid + i * 256;        // 512 chunks of 8 bf16
      int row  = l >> 2;
      int col8 = (l & 3) * 8;
      *(uint4*)&As[row][col8] =
          *(const uint4*)&A[(size_t)(m0 + row) * K + k0 + col8];
      *(uint4*)&Bs[row][col8] =
          *(const uint4*)&W[(size_t)(n0 + row) * K + k0 + col8];
    }
    // prefetch next K-slab while this one is consumed (global_prefetch_b8)
    if (k0 + 32 < K) {
#pragma unroll
      for (int i = 0; i < 2; ++i) {
        int l    = tid + i * 256;
        int row  = l >> 2;
        int col8 = (l & 3) * 8;
        __builtin_prefetch(&A[(size_t)(m0 + row) * K + k0 + 32 + col8], 0, 1);
        __builtin_prefetch(&W[(size_t)(n0 + row) * K + k0 + 32 + col8], 0, 1);
      }
    }
    __syncthreads();

    v16bf af[2], bfr[4];
#pragma unroll
    for (int mi = 0; mi < 2; ++mi) {
      const __bf16* p = &As[wm + mi * 16 + laneLo][hi * 8];
#pragma unroll
      for (int e = 0; e < 8; ++e) { af[mi][e] = p[e]; af[mi][8 + e] = p[16 + e]; }
    }
#pragma unroll
    for (int ni = 0; ni < 4; ++ni) {
      const __bf16* p = &Bs[wn + ni * 16 + laneLo][hi * 8];
#pragma unroll
      for (int e = 0; e < 8; ++e) { bfr[ni][e] = p[e]; bfr[ni][8 + e] = p[16 + e]; }
    }
#pragma unroll
    for (int mi = 0; mi < 2; ++mi)
#pragma unroll
      for (int ni = 0; ni < 4; ++ni)
        acc[mi][ni] = wmma_bf16(af[mi], bfr[ni], acc[mi][ni]);
    __syncthreads();
  }

#pragma unroll
  for (int mi = 0; mi < 2; ++mi)
#pragma unroll
    for (int ni = 0; ni < 4; ++ni)
#pragma unroll
      for (int r = 0; r < 8; ++r) {
        int row = m0 + wm + mi * 16 + r + hi * 8;
        int col = n0 + wn + ni * 16 + laneLo;
        float v = acc[mi][ni][r];
        if constexpr (OUT_BF16)
          ((__bf16*)Cout)[(size_t)row * N + col] = to_bf16(v);
        else
          ((float*)Cout)[(size_t)row * N + col] = v;
      }
}

// ---------------------------------------------------------------------------
// RoPE + relayout: qkv[B,T,3,H,Dh] (bf16, col o = s*2048 + h*128 + d)
// -> q_r/k_r/v_r [B,H,T,Dh] bf16, rotate-half RoPE on q,k (half = 64).
// One thread per (b,h,t,d), d in [0,64).
// ---------------------------------------------------------------------------
__global__ __launch_bounds__(256)
void rope_relayout(const __bf16* __restrict__ qkv,
                   __bf16* __restrict__ qr, __bf16* __restrict__ kr,
                   __bf16* __restrict__ vr, int B, int T, int H) {
  const int Dh = 128, half = 64;
  size_t i = (size_t)blockIdx.x * blockDim.x + threadIdx.x;
  size_t total = (size_t)B * H * T * half;
  if (i >= total) return;
  int d = (int)(i % half);
  int t = (int)((i / half) % T);
  int h = (int)((i / ((size_t)half * T)) % H);
  int b = (int)(i / ((size_t)half * T * H));

  const int C3 = 3 * H * Dh;
  const __bf16* row = qkv + ((size_t)b * T + t) * C3;
  int base = h * Dh + d;
  float q1 = bf16_to_f32(row[base]);
  float q2 = bf16_to_f32(row[base + half]);
  float k1 = bf16_to_f32(row[H * Dh + base]);
  float k2 = bf16_to_f32(row[H * Dh + base + half]);
  float v1 = bf16_to_f32(row[2 * H * Dh + base]);
  float v2 = bf16_to_f32(row[2 * H * Dh + base + half]);

  // inv_d = 10000^(-d/64) = exp(d * -ln(10000)/64)
  float ang = (float)t * __expf((float)d * -0.14391156831212787f);
  float s = __sinf(ang), c = __cosf(ang);

  size_t o = (((size_t)b * H + h) * T + t) * Dh + d;
  qr[o]        = to_bf16(q1 * c - q2 * s);
  qr[o + half] = to_bf16(q2 * c + q1 * s);
  kr[o]        = to_bf16(k1 * c - k2 * s);
  kr[o + half] = to_bf16(k2 * c + k1 * s);
  vr[o]        = to_bf16(v1);
  vr[o + half] = to_bf16(v2);
}

// ---------------------------------------------------------------------------
// Flash attention: block = (b, h, 128 q-rows), 8 waves x 16 rows each.
// K tile row-major in LDS (B-frags contiguous); V tile stored TRANSPOSED
// (Vt[dh][kk]) so PV B-frags are also contiguous ds_load_b128.
// Online softmax; PV accumulated in f32 frags; bf16 out to attn[B,T,H*Dh].
// ---------------------------------------------------------------------------
__global__ __launch_bounds__(256)
void flash_attn(const __bf16* __restrict__ Q, const __bf16* __restrict__ Kg,
                const __bf16* __restrict__ Vg, __bf16* __restrict__ Og,
                int T, int H) {
  const int Dh = 128;
  __shared__ __align__(16) __bf16 Ks[64][Dh + 8];   // 17408 B
  __shared__ __align__(16) __bf16 Vt[128][72];      // 18432 B (transposed V)
  __shared__ __align__(16) __bf16 Ps[8][16][72];    // 18432 B per-wave P

  const int tid    = threadIdx.x;
  const int lane   = tid & 31;
  const int wave   = tid >> 5;
  const int laneLo = lane & 15;
  const int hi     = lane >> 4;
  const int q0 = blockIdx.x * 128;
  const int h  = blockIdx.y;
  const int b  = blockIdx.z;
  const size_t headOff = ((size_t)b * H + h) * (size_t)T * Dh;
  const int qw = q0 + wave * 16;

  // Q fragments (A-layout), 4 chunks of K-depth 32.
  v16bf qa[4];
  {
    const __bf16* qrow = Q + headOff + (size_t)(qw + laneLo) * Dh;
#pragma unroll
    for (int dc = 0; dc < 4; ++dc)
#pragma unroll
      for (int e = 0; e < 8; ++e) {
        qa[dc][e]     = qrow[dc * 32 + hi * 8 + e];
        qa[dc][8 + e] = qrow[dc * 32 + hi * 8 + 16 + e];
      }
  }

  v8f o[8];
#pragma unroll
  for (int dc = 0; dc < 8; ++dc)
    o[dc] = (v8f){0.f, 0.f, 0.f, 0.f, 0.f, 0.f, 0.f, 0.f};
  float mrow[8], lrow[8];
#pragma unroll
  for (int r = 0; r < 8; ++r) { mrow[r] = -3.0e38f; lrow[r] = 0.f; }

  const float scale = 0.08838834764831845f;  // 1/sqrt(128)
  const int ktiles = (q0 + 128) / 64;

  for (int kt = 0; kt < ktiles; ++kt) {
    const int k0 = kt * 64;
    __syncthreads();  // previous iteration's LDS consumers done
#pragma unroll
    for (int i = 0; i < 4; ++i) {
      int l    = tid + i * 256;           // 1024 chunks of 8 bf16
      int row  = l >> 4;
      int col8 = (l & 15) * 8;
      *(uint4*)&Ks[row][col8] =
          *(const uint4*)&Kg[headOff + (size_t)(k0 + row) * Dh + col8];
      // V: load 8 contiguous dh values of row kk, scatter transposed.
      uint4 vv = *(const uint4*)&Vg[headOff + (size_t)(k0 + row) * Dh + col8];
      __bf16 ve[8];
      *(uint4*)ve = vv;
#pragma unroll
      for (int j = 0; j < 8; ++j) Vt[col8 + j][row] = ve[j];
    }
    // prefetch next K/V tile (global_prefetch_b8)
    if (kt + 1 < ktiles) {
      const int kn = k0 + 64;
#pragma unroll
      for (int i = 0; i < 4; ++i) {
        int l    = tid + i * 256;
        int row  = l >> 4;
        int col8 = (l & 15) * 8;
        __builtin_prefetch(&Kg[headOff + (size_t)(kn + row) * Dh + col8], 0, 1);
        __builtin_prefetch(&Vg[headOff + (size_t)(kn + row) * Dh + col8], 0, 1);
      }
    }
    __syncthreads();

    const bool active = (k0 <= qw + 15);  // wave-uniform causal skip
    float cscale[8];
    if (active) {
      // scores: S = Q x K^T (16 q-rows x 64 k-cols), K-depth 128.
      v8f s[4];
#pragma unroll
      for (int n16 = 0; n16 < 4; ++n16) {
        v8f cacc = (v8f){0.f, 0.f, 0.f, 0.f, 0.f, 0.f, 0.f, 0.f};
#pragma unroll
        for (int dc = 0; dc < 4; ++dc) {
          v16bf bk;
          const __bf16* p = &Ks[n16 * 16 + laneLo][dc * 32 + hi * 8];
#pragma unroll
          for (int e = 0; e < 8; ++e) { bk[e] = p[e]; bk[8 + e] = p[16 + e]; }
          cacc = wmma_bf16(qa[dc], bk, cacc);
        }
        s[n16] = cacc;
      }
      // scale + causal mask
#pragma unroll
      for (int n16 = 0; n16 < 4; ++n16)
#pragma unroll
        for (int r = 0; r < 8; ++r) {
          int qrow = qw + r + hi * 8;
          int kcol = k0 + n16 * 16 + laneLo;
          float v = s[n16][r] * scale;
          s[n16][r] = (kcol > qrow) ? -1.0e30f : v;
        }
      // online softmax (row lives across 16 lanes of the half-wave)
#pragma unroll
      for (int r = 0; r < 8; ++r) {
        float tmax = s[0][r];
#pragma unroll
        for (int n16 = 1; n16 < 4; ++n16) tmax = fmaxf(tmax, s[n16][r]);
#pragma unroll
        for (int m = 1; m < 16; m <<= 1)
          tmax = fmaxf(tmax, __shfl_xor(tmax, m, 32));
        float mnew = fmaxf(mrow[r], tmax);
        float cs = __expf(mrow[r] - mnew);
        mrow[r] = mnew;
        float rs = 0.f;
#pragma unroll
        for (int n16 = 0; n16 < 4; ++n16) {
          float p = __expf(s[n16][r] - mnew);
          s[n16][r] = p;
          rs += p;
        }
#pragma unroll
        for (int m = 1; m < 16; m <<= 1) rs += __shfl_xor(rs, m, 32);
        lrow[r] = lrow[r] * cs + rs;
        cscale[r] = cs;
      }
      // rescale O, stage P into LDS (C-layout -> A-layout transpose)
#pragma unroll
      for (int dc = 0; dc < 8; ++dc)
#pragma unroll
        for (int r = 0; r < 8; ++r) o[dc][r] *= cscale[r];
#pragma unroll
      for (int n16 = 0; n16 < 4; ++n16)
#pragma unroll
        for (int r = 0; r < 8; ++r)
          Ps[wave][r + hi * 8][n16 * 16 + laneLo] = to_bf16(s[n16][r]);
    }
    __syncthreads();
    if (active) {
      v16bf pa[2];
#pragma unroll
      for (int kh = 0; kh < 2; ++kh) {
        const __bf16* pp = &Ps[wave][laneLo][kh * 32 + hi * 8];
#pragma unroll
        for (int e = 0; e < 8; ++e) { pa[kh][e] = pp[e]; pa[kh][8 + e] = pp[16 + e]; }
      }
#pragma unroll
      for (int dc = 0; dc < 8; ++dc)
#pragma unroll
        for (int kh = 0; kh < 2; ++kh) {
          // B-frag from transposed V: two runs of 8 bf16 per lane (k = 8h+e, 16+8h+e).
          v16bf bv;
          const __bf16* vp = &Vt[dc * 16 + laneLo][kh * 32 + hi * 8];
#pragma unroll
          for (int e = 0; e < 8; ++e) { bv[e] = vp[e]; bv[8 + e] = vp[16 + e]; }
          o[dc] = wmma_bf16(pa[kh], bv, o[dc]);
        }
    }
  }

  // epilogue: O / l, write bf16 to [B, T, H*Dh]
#pragma unroll
  for (int dc = 0; dc < 8; ++dc)
#pragma unroll
    for (int r = 0; r < 8; ++r) {
      int qrow = qw + r + hi * 8;
      int col = h * Dh + dc * 16 + laneLo;
      float v = o[dc][r] / lrow[r];
      Og[((size_t)b * T + qrow) * (size_t)(H * Dh) + col] = to_bf16(v);
    }
}

// ---------------------------------------------------------------------------
// Host-side launch
// ---------------------------------------------------------------------------
extern "C" void kernel_launch(void* const* d_in, const int* in_sizes, int n_in,
                              void* d_out, int out_size, void* d_ws, size_t ws_size,
                              hipStream_t stream) {
  const int B = 2, T = 2048, C = 2048, H = 16, Dh = 128;
  const float* x      = (const float*)d_in[0];  // [B,T,C]
  const float* w_qkv  = (const float*)d_in[1];  // [3C,C]
  const float* w_proj = (const float*)d_in[2];  // [C,C]
  float* out = (float*)d_out;                   // [B,T,C] fp32

  const size_t nX    = (size_t)B * T * C;       // 8,388,608
  const size_t nWqkv = (size_t)3 * C * C;       // 12,582,912
  const size_t nWprj = (size_t)C * C;           // 4,194,304
  const size_t nQKV  = (size_t)B * T * 3 * C;   // 25,165,824

  char* p = (char*)d_ws;
  __bf16* x_bf    = (__bf16*)p; p += nX * 2;
  __bf16* wqkv_bf = (__bf16*)p; p += nWqkv * 2;
  __bf16* wprj_bf = (__bf16*)p; p += nWprj * 2;
  __bf16* qkv_bf  = (__bf16*)p; p += nQKV * 2;
  __bf16* q_r     = (__bf16*)p; p += nX * 2;
  __bf16* k_r     = (__bf16*)p; p += nX * 2;
  __bf16* v_r     = (__bf16*)p; p += nX * 2;
  __bf16* attn_bf = (__bf16*)p; p += nX * 2;    // total ~160 MB

  // 1) convert inputs to bf16
  cvt_f32_bf16<<<4096, 256, 0, stream>>>(x, x_bf, nX);
  cvt_f32_bf16<<<4096, 256, 0, stream>>>(w_qkv, wqkv_bf, nWqkv);
  cvt_f32_bf16<<<4096, 256, 0, stream>>>(w_proj, wprj_bf, nWprj);

  // 2) qkv = x @ w_qkv^T : M=B*T=4096, N=3C=6144, K=C=2048 (bf16 out)
  {
    dim3 grid(6144 / 128, 4096 / 128);
    gemm_xwt<true><<<grid, 256, 0, stream>>>(x_bf, wqkv_bf, (void*)qkv_bf,
                                             4096, 6144, 2048);
  }

  // 3) RoPE + relayout to [B,H,T,Dh]
  {
    size_t total = (size_t)B * H * T * (Dh / 2);  // 4,194,304
    rope_relayout<<<(unsigned)((total + 255) / 256), 256, 0, stream>>>(
        qkv_bf, q_r, k_r, v_r, B, T, H);
  }

  // 4) flash attention -> attn_bf [B,T,C]
  {
    dim3 grid(T / 128, H, B);
    flash_attn<<<grid, 256, 0, stream>>>(q_r, k_r, v_r, attn_bf, T, H);
  }

  // 5) out = attn @ w_proj^T : M=4096, N=2048, K=2048 (fp32 out)
  {
    dim3 grid(2048 / 128, 4096 / 128);
    gemm_xwt<false><<<grid, 256, 0, stream>>>(attn_bf, wprj_bf, (void*)out,
                                              4096, 2048, 2048);
  }
}
